// SelfAttention_31421980737894
// MI455X (gfx1250) — matrix operand order repaired and checked by reading, not compiled, not run on hardware
//
#include <hip/hip_runtime.h>

// ---------------- types ----------------
typedef __bf16 bf16_t;
typedef bf16_t v8bf  __attribute__((ext_vector_type(8)));
typedef bf16_t v16bf __attribute__((ext_vector_type(16)));
typedef float  v8f   __attribute__((ext_vector_type(8)));

#define B_ 8
#define C_ 256
#define N_ 4096   // 64*64
#define NJ (N_ / 32)

// padded LDS row strides (bf16 units; all keep 16B alignment, avoid bank conflicts)
#define KROW 264   // k tile: 32 rows x 256 cols
#define VROW 40    // v tile: 256 rows x 32 cols
#define PROW 40    // P staging: 16 rows x 32 cols

// ---------------- WMMA helpers ----------------
__device__ inline v8f wmma_bf16(v16bf a, v16bf b, v8f c) {
  return __builtin_amdgcn_wmma_f32_16x16x32_bf16(
      /*neg_a=*/false, a, /*neg_b=*/false, b,
      /*c_mod=*/(short)0, c, /*reuse_a=*/false, /*reuse_b=*/false);
}

// A-operand (16x32 MxK): lane<16 -> M=lane, K=kBase+[0..15];
// lane>=16 -> M=lane-16, K=kBase+[16..31]
__device__ inline v16bf load_a_bf16(const bf16_t* rowPtr, int colBase) {
  v8bf lo = *reinterpret_cast<const v8bf*>(rowPtr + colBase);
  v8bf hi = *reinterpret_cast<const v8bf*>(rowPtr + colBase + 8);
  v16bf a;
#pragma unroll
  for (int i = 0; i < 8; ++i) { a[i] = lo[i]; a[8 + i] = hi[i]; }
  return a;
}

// B-operand (32x16 KxN): each lane reads 16 contiguous bf16
__device__ inline v16bf load_b_bf16(const bf16_t* p) {
  v8bf lo = *reinterpret_cast<const v8bf*>(p);
  v8bf hi = *reinterpret_cast<const v8bf*>(p + 8);
  v16bf b;
#pragma unroll
  for (int i = 0; i < 8; ++i) { b[i] = lo[i]; b[8 + i] = hi[i]; }
  return b;
}

// B-operand from fp32 (weights): 16 contiguous floats -> bf16
__device__ inline v16bf load_b_f32(const float* p) {
  v16bf b;
#pragma unroll
  for (int i = 0; i < 16; ++i) b[i] = (bf16_t)p[i];
  return b;
}

// intra-16-lane reductions (wave32; masks stay inside each 16-lane group)
__device__ inline float rowmax16(float x) {
#pragma unroll
  for (int m = 1; m < 16; m <<= 1) x = fmaxf(x, __shfl_xor(x, m, 32));
  return x;
}
__device__ inline float rowsum16(float x) {
#pragma unroll
  for (int m = 1; m < 16; m <<= 1) x += __shfl_xor(x, m, 32);
  return x;
}

// CDNA5 async global->LDS copy, 16B per lane, tracked by ASYNCcnt.
__device__ inline void async_b128(bf16_t* lds_p, const bf16_t* g_p) {
  __attribute__((address_space(3))) bf16_t* lp =
      (__attribute__((address_space(3))) bf16_t*)lds_p;
  asm volatile("global_load_async_to_lds_b128 %0, %1, off"
               :: "v"(lp), "v"(g_p) : "memory");
}

// ---------------- kernel 1: x (B,C,N) f32 -> xT (B,N,C) bf16 ----------------
__global__ __launch_bounds__(256) void transpose_cast_kernel(
    const float* __restrict__ x, bf16_t* __restrict__ xT) {
  __shared__ float tile[32][33];
  const int b  = blockIdx.z;
  const int n0 = blockIdx.x * 32;
  const int c0 = blockIdx.y * 32;
  const int tx = threadIdx.x;
  const int ty = threadIdx.y;
  const float* xp = x + ((size_t)b * C_ + c0) * (size_t)N_ + n0;
#pragma unroll
  for (int k = 0; k < 4; ++k)
    tile[ty + 8 * k][tx] = xp[(size_t)(ty + 8 * k) * N_ + tx];
  __syncthreads();
  bf16_t* xo = xT + ((size_t)b * N_ + n0) * (size_t)C_ + c0;
#pragma unroll
  for (int k = 0; k < 4; ++k)
    xo[(size_t)(ty + 8 * k) * C_ + tx] = (bf16_t)tile[tx][ty + 8 * k];
}

// ---------------- kernel 2: q,k projections ----------------
__global__ __launch_bounds__(128) void proj_qk_kernel(
    const bf16_t* __restrict__ xT,
    const float* __restrict__ Wq, const float* __restrict__ bq,
    const float* __restrict__ Wk, const float* __restrict__ bk,
    bf16_t* __restrict__ qT, bf16_t* __restrict__ kT) {
  const int lane = threadIdx.x & 31;
  const int wid  = threadIdx.x >> 5;
  const int w    = blockIdx.x * 4 + wid;
  const int b    = w >> 12;
  const int rem  = w & 4095;
  const int n0   = (rem >> 4) * 16;
  const int c0   = (rem & 15) * 16;
  const int ln   = lane & 15;
  const int half = lane >> 4;

  const bf16_t* arow = xT + ((size_t)b * N_ + n0 + ln) * C_;
  v8f aq = {0.f,0.f,0.f,0.f,0.f,0.f,0.f,0.f};
  v8f ak = {0.f,0.f,0.f,0.f,0.f,0.f,0.f,0.f};
#pragma unroll
  for (int kc = 0; kc < 8; ++kc) {
    v16bf a = load_a_bf16(arow, kc * 32 + half * 16);
    const float* wq = Wq + (size_t)(c0 + ln) * C_ + kc * 32 + half * 16;
    const float* wk = Wk + (size_t)(c0 + ln) * C_ + kc * 32 + half * 16;
    aq = wmma_bf16(a, load_b_f32(wq), aq);
    ak = wmma_bf16(a, load_b_f32(wk), ak);
  }
  const float biasq = bq[c0 + ln];
  const float biask = bk[c0 + ln];
#pragma unroll
  for (int r = 0; r < 8; ++r) {
    const int row = n0 + r + 8 * half;
    const size_t idx = ((size_t)b * N_ + row) * C_ + c0 + ln;
    qT[idx] = (bf16_t)(aq[r] + biasq);
    kT[idx] = (bf16_t)(ak[r] + biask);
  }
}

// ---------------- kernel 3: v projection ----------------
__global__ __launch_bounds__(128) void proj_v_kernel(
    const bf16_t* __restrict__ xT,
    const float* __restrict__ Wv, const float* __restrict__ bv,
    bf16_t* __restrict__ vm) {
  const int lane = threadIdx.x & 31;
  const int wid  = threadIdx.x >> 5;
  const int w    = blockIdx.x * 4 + wid;
  const int b    = w >> 12;
  const int rem  = w & 4095;
  const int n0   = (rem >> 4) * 16;
  const int c0   = (rem & 15) * 16;
  const int ln   = lane & 15;
  const int half = lane >> 4;

  const float* wrow = Wv + (size_t)(c0 + ln) * C_;
  v8f acc = {0.f,0.f,0.f,0.f,0.f,0.f,0.f,0.f};
#pragma unroll
  for (int kc = 0; kc < 8; ++kc) {
    const int cb = kc * 32 + half * 16;
    v16bf a;
#pragma unroll
    for (int i = 0; i < 8; ++i) {
      a[i]     = (bf16_t)wrow[cb + i];
      a[8 + i] = (bf16_t)wrow[cb + 8 + i];
    }
    const bf16_t* xrow = xT + ((size_t)b * N_ + n0 + ln) * C_ + kc * 32 + half * 16;
    acc = wmma_bf16(a, load_b_bf16(xrow), acc);
  }
#pragma unroll
  for (int r = 0; r < 8; ++r) {
    const int co = c0 + r + 8 * half;
    vm[((size_t)b * C_ + co) * N_ + n0 + ln] = (bf16_t)(acc[r] + bv[co]);
  }
}

// ---------------- kernel 4: fused flash attention + residual ----------------
// 8 waves / WG; each wave owns 16 query rows; all waves share async-staged
// k/v j-tiles in LDS (double-buffered), cutting L2 traffic 8x.
__global__ __launch_bounds__(256) void attn_kernel(
    const bf16_t* __restrict__ qT, const bf16_t* __restrict__ kT,
    const bf16_t* __restrict__ vm, const float* __restrict__ x,
    const float* __restrict__ gamma, float* __restrict__ out) {
  __shared__ bf16_t kbuf[2][32 * KROW];    // 33 KB total
  __shared__ bf16_t vbuf[2][256 * VROW];   // 40 KB total
  __shared__ bf16_t Pst[8][16 * PROW];     // 10 KB total

  const int tid  = threadIdx.x;
  const int lane = tid & 31;
  const int wid  = tid >> 5;
  const int w    = blockIdx.x * 8 + wid;   // 0 .. B*256-1 (8 waves share batch b)
  const int b    = w >> 8;
  const int i0   = (w & 255) * 16;
  const int ln   = lane & 15;
  const int half = lane >> 4;

  const bf16_t* kbase = kT + (size_t)b * N_ * C_;
  const bf16_t* vbase = vm + (size_t)b * C_ * N_;

  // cooperative async stage of one j-block (32 wide): k 16KB + v 16KB
  auto stage = [&](int j0, int buf) {
    const bf16_t* kg = kbase + (size_t)j0 * C_;   // 32 contiguous rows of 512B
    bf16_t* kd = &kbuf[buf][0];
#pragma unroll
    for (int i = 0; i < 4; ++i) {
      const int ch = tid + i * 256;               // 1024 chunks of 16B
      const int r = ch >> 5, cc = ch & 31;
      async_b128(kd + r * KROW + cc * 8, kg + r * C_ + cc * 8);
    }
    const bf16_t* vg = vbase + j0;                // row c: 64B segment, stride N
    bf16_t* vd = &vbuf[buf][0];
#pragma unroll
    for (int i = 0; i < 4; ++i)
      async_b128(vd + tid * VROW + i * 8, vg + (size_t)tid * N_ + i * 8);
  };

  // preload q A-chunks: 16 rows x 256 cols in 8 K-chunks of 32
  const bf16_t* qrow = qT + ((size_t)b * N_ + i0 + ln) * C_;
  v16bf qa[8];
#pragma unroll
  for (int kc = 0; kc < 8; ++kc) qa[kc] = load_a_bf16(qrow, kc * 32 + half * 16);

  v8f acc[16];
#pragma unroll
  for (int t = 0; t < 16; ++t)
    acc[t] = (v8f){0.f,0.f,0.f,0.f,0.f,0.f,0.f,0.f};
  float mrun[8], lrun[8];
#pragma unroll
  for (int r = 0; r < 8; ++r) { mrun[r] = -3.0e38f; lrun[r] = 0.f; }

  stage(0, 0);   // prologue: fill buffer 0

  for (int jb = 0; jb < NJ; ++jb) {
    const int cur = jb & 1;
    if (jb + 1 < NJ) {
      stage((jb + 1) * 32, cur ^ 1);                       // overlap next DMA
      asm volatile("s_wait_asynccnt 0x8" ::: "memory");    // this wave's prior 8 done
    } else {
      asm volatile("s_wait_asynccnt 0x0" ::: "memory");
    }
    __syncthreads();                                       // all waves' tiles visible

    const bf16_t* kb = &kbuf[cur][0];
    const bf16_t* vb = &vbuf[cur][0];

    // ---- energy tiles from LDS ----
    v8f e0 = (v8f){0.f,0.f,0.f,0.f,0.f,0.f,0.f,0.f};
    v8f e1 = (v8f){0.f,0.f,0.f,0.f,0.f,0.f,0.f,0.f};
#pragma unroll
    for (int kc = 0; kc < 8; ++kc) {
      const int cb = kc * 32 + half * 16;
      e0 = wmma_bf16(qa[kc], load_b_bf16(kb + (size_t)ln * KROW + cb), e0);
      e1 = wmma_bf16(qa[kc], load_b_bf16(kb + (size_t)(16 + ln) * KROW + cb), e1);
    }

    // ---- online softmax (row stats replicated across 16-lane groups) ----
    float sc[8];
#pragma unroll
    for (int r = 0; r < 8; ++r) {
      float cur_m = rowmax16(fmaxf(e0[r], e1[r]));
      const float mn = fmaxf(mrun[r], cur_m);
      sc[r] = __expf(mrun[r] - mn);
      const float p0 = __expf(e0[r] - mn);
      const float p1 = __expf(e1[r] - mn);
      lrun[r] = lrun[r] * sc[r] + rowsum16(p0 + p1);
      mrun[r] = mn;
      const int row = r + 8 * half;                 // D-layout: M in VGPR idx
      Pst[wid][row * PROW + ln]      = (bf16_t)p0;  // j cols 0..15
      Pst[wid][row * PROW + 16 + ln] = (bf16_t)p1;  // j cols 16..31
    }
#pragma unroll
    for (int t = 0; t < 16; ++t)
#pragma unroll
      for (int r = 0; r < 8; ++r) acc[t][r] *= sc[r];

    // wave-local LDS RAW fence (cross-lane transpose through LDS)
    asm volatile("s_wait_dscnt 0x0" ::: "memory");

    // re-read P in A-operand layout (already bf16)
    const bf16_t* pr = &Pst[wid][ln * PROW + half * 16];
    v8bf plo = *reinterpret_cast<const v8bf*>(pr);
    v8bf phi = *reinterpret_cast<const v8bf*>(pr + 8);
    v16bf pa;
#pragma unroll
    for (int i = 0; i < 8; ++i) { pa[i] = plo[i]; pa[8 + i] = phi[i]; }

    // ---- PV from LDS: 16 c-tiles, K=32 ----
#pragma unroll
    for (int t = 0; t < 16; ++t)
      acc[t] = wmma_bf16(pa, load_b_bf16(vb + (size_t)(t * 16 + ln) * VROW + half * 16),
                         acc[t]);

    __syncthreads();   // all waves done with this buffer before it is re-staged
  }

  // ---- finalize: softmax/sqrt(N), gamma*out + x ----
  const float g = gamma[0];
  float inv[8];
#pragma unroll
  for (int r = 0; r < 8; ++r) inv[r] = 1.0f / (lrun[r] * 64.0f);  // sqrt(4096)=64
#pragma unroll
  for (int t = 0; t < 16; ++t) {
#pragma unroll
    for (int r = 0; r < 8; ++r) {
      const int ci = t * 16 + ln;
      const int ii = i0 + r + 8 * half;
      const size_t idx = ((size_t)b * C_ + ci) * (size_t)N_ + ii;
      out[idx] = g * (acc[t][r] * inv[r]) + x[idx];
    }
  }
}

// ---------------- launcher ----------------
extern "C" void kernel_launch(void* const* d_in, const int* in_sizes, int n_in,
                              void* d_out, int out_size, void* d_ws, size_t ws_size,
                              hipStream_t stream) {
  const float* x     = (const float*)d_in[0];
  const float* Wq    = (const float*)d_in[1];
  const float* bq    = (const float*)d_in[2];
  const float* Wk    = (const float*)d_in[3];
  const float* bk    = (const float*)d_in[4];
  const float* Wv    = (const float*)d_in[5];
  const float* bv    = (const float*)d_in[6];
  const float* gamma = (const float*)d_in[7];
  float* out = (float*)d_out;

  char* ws = (char*)d_ws;
  const size_t szT = (size_t)B_ * N_ * C_ * sizeof(bf16_t);   // 16 MB each
  bf16_t* xT = (bf16_t*)(ws);
  bf16_t* qT = (bf16_t*)(ws + szT);
  bf16_t* kT = (bf16_t*)(ws + 2 * szT);
  bf16_t* vm = (bf16_t*)(ws + 3 * szT);

  transpose_cast_kernel<<<dim3(N_ / 32, C_ / 32, B_), dim3(32, 8), 0, stream>>>(x, xT);
  proj_qk_kernel<<<(B_ * 256 * 16) / 4, 128, 0, stream>>>(xT, Wq, bq, Wk, bk, qT, kT);
  proj_v_kernel<<<(B_ * 256 * 16) / 4, 128, 0, stream>>>(xT, Wv, bv, vm);
  attn_kernel<<<(B_ * 256) / 8, 256, 0, stream>>>(qT, kT, vm, x, gamma, out);
}
